// BottomBlock_63350767616771
// MI455X (gfx1250) — matrix operand order repaired and checked by reading, not compiled, not run on hardware
//
#include <hip/hip_runtime.h>
#include <cstdint>
#include <cstddef>

#define BB     2
#define TT     2048
#define DIMM   2048
#define HEADS  16
#define HD     128
#define KSZ    4
#define CHUNK  64
#define INTER  8192
#define HK     (HEADS*HD)
#define BT     (BB*TT)
#define NCHUNK (TT/CHUNK)

typedef __attribute__((ext_vector_type(16))) __bf16 v16bf;
typedef __attribute__((ext_vector_type(8)))  float  v8f;
typedef __attribute__((ext_vector_type(4)))  int    v4i_t;
typedef __attribute__((ext_vector_type(8)))  int    v8i_t;
typedef __attribute__((ext_vector_type(4)))  unsigned int v4u_t;

// ---- CDNA5 feature probes (device pass only; host pass takes fallbacks) ----
#if defined(__gfx1250__) && __has_builtin(__builtin_amdgcn_global_load_async_to_lds_b128) && __has_builtin(__builtin_amdgcn_s_wait_asynccnt)
#define HAVE_ASYNC 1
#else
#define HAVE_ASYNC 0
#endif

#if defined(__gfx1250__) && __has_builtin(__builtin_amdgcn_tensor_load_to_lds) && __has_builtin(__builtin_amdgcn_s_wait_tensorcnt)
#define HAVE_TDM 1
#else
#define HAVE_TDM 0
#endif

#if __has_include(<hip/amd_detail/amd_gfx1250_TDM.h>)
#define TDM_SIX_ARGS 1
#else
#define TDM_SIX_ARGS 0
#endif

#define GAS __attribute__((address_space(1)))
#define LAS __attribute__((address_space(3)))

union Frag16 {
  v16bf v;
  uint4 q[2];
  unsigned short u[16];
};

__device__ __forceinline__ unsigned short f2bf(float f) {
  union { float f; unsigned u; } c; c.f = f;
  unsigned r = c.u + 0x7fffu + ((c.u >> 16) & 1u);
  return (unsigned short)(r >> 16);
}
__device__ __forceinline__ float bf2f(unsigned short h) {
  union { float f; unsigned u; } c; c.u = ((unsigned)h) << 16; return c.f;
}

__device__ __forceinline__ v8f wmma_bf16(v16bf a, v16bf b, v8f c) {
  return __builtin_amdgcn_wmma_f32_16x16x32_bf16(false, a, false, b, (short)0, c, false, false);
}

// ---- WMMA fragment loaders (contiguous k order: element s of lane half h holds k = 16h + s) ----
__device__ __forceinline__ v16bf fragA_rm(const unsigned short* base, int ld) {
  int lane = threadIdx.x & 31;
  int r = lane & 15;
  int c0 = (lane >> 4) * 16;
  Frag16 f;
  f.q[0] = *(const uint4*)(base + r * ld + c0);
  f.q[1] = *(const uint4*)(base + r * ld + c0 + 8);
  return f.v;
}
__device__ __forceinline__ v16bf fragBt_rm(const unsigned short* base, int ld) {
  int lane = threadIdx.x & 31;
  int r = lane & 15;
  int c0 = (lane >> 4) * 16;
  Frag16 f;
  f.q[0] = *(const uint4*)(base + r * ld + c0);
  f.q[1] = *(const uint4*)(base + r * ld + c0 + 8);
  return f.v;
}
__device__ __forceinline__ v16bf fragB_kmajor(const unsigned short* base, int ld, int n0) {
  int lane = threadIdx.x & 31;
  int nn = n0 + (lane & 15);
  int k0 = (lane >> 4) * 16;
  Frag16 f;
#pragma unroll
  for (int t = 0; t < 16; ++t) f.u[t] = base[(size_t)(k0 + t) * ld + nn];
  return f.v;
}
__device__ __forceinline__ v16bf fragA_gatherT(const unsigned short* base, int ld, int m0) {
  int lane = threadIdx.x & 31;
  int m = m0 + (lane & 15);
  int c0 = (lane >> 4) * 16;
  Frag16 f;
#pragma unroll
  for (int t = 0; t < 8; ++t) {
    f.u[t]     = base[(size_t)(c0 + t) * ld + m];
    f.u[8 + t] = base[(size_t)(c0 + 8 + t) * ld + m];
  }
  return f.v;
}

// ---- Tensor Data Mover: 2D bf16 tile (tile_w units wide, tile_h rows) ----
#if HAVE_TDM
__device__ __forceinline__ unsigned lds_off_of(const void* p) {
  return (unsigned)(unsigned long long)(LAS const void*)p;
}
__device__ __forceinline__ void tdm_load_2d_bf16(const void* gsrc, const void* ldst,
                                                 unsigned tile_w, unsigned tile_h,
                                                 unsigned row_stride_units) {
  unsigned long long ga = (unsigned long long)gsrc;
  unsigned td0 = 1u << 20, td1 = 1u << 20;  // generous tensor dims (no OOB clipping needed)
  unsigned w[12];
  // group 0 (D# 128b): count=1 | lds_addr | global_addr | type=2
  w[0] = 1u;
  w[1] = lds_off_of(ldst);
  w[2] = (unsigned)(ga & 0xffffffffu);
  w[3] = (unsigned)((ga >> 32) & 0x01ffffffu) | 0x80000000u;
  // group 1 (256b): data_size=2B, tensor_dim0/1, tile_dim0/1, dim0 stride
  w[4]  = 1u << 16;
  w[5]  = (td0 & 0xffffu) << 16;
  w[6]  = (td0 >> 16) | ((td1 & 0xffffu) << 16);
  w[7]  = (td1 >> 16) | (tile_w << 16);
  w[8]  = tile_h;
  w[9]  = 0u;
  w[10] = row_stride_units;
  w[11] = 0u;
#pragma unroll
  for (int i = 0; i < 12; ++i) w[i] = (unsigned)__builtin_amdgcn_readfirstlane((int)w[i]);
  v4u_t g0 = { w[0], w[1], w[2], w[3] };
  v8i_t g1 = { (int)w[4], (int)w[5], (int)w[6], (int)w[7],
               (int)w[8], (int)w[9], (int)w[10], (int)w[11] };
  v4i_t z4 = { 0, 0, 0, 0 };
#if TDM_SIX_ARGS
  v8i_t z8 = { 0, 0, 0, 0, 0, 0, 0, 0 };
  __builtin_amdgcn_tensor_load_to_lds(g0, g1, z4, z4, z8, 0);
#else
  __builtin_amdgcn_tensor_load_to_lds(g0, g1, z4, z4, 0);
#endif
}
#endif

// -------------------- elementwise / small kernels --------------------
__global__ void cvt_f32_bf16_kernel(const float* __restrict__ s, unsigned short* __restrict__ d, size_t n) {
  size_t i = (size_t)blockIdx.x * blockDim.x + threadIdx.x;
  if (i < n) d[i] = f2bf(s[i]);
}

__global__ void rmsnorm_kernel(const float* __restrict__ x, const float* __restrict__ w,
                               unsigned short* __restrict__ out, int cols) {
  __shared__ float red[256];
  size_t row = blockIdx.x;
  const float* xr = x + row * (size_t)cols;
  int tid = threadIdx.x;
  float ss = 0.f;
  for (int c = tid; c < cols; c += 256) { float v = xr[c]; ss += v * v; }
  red[tid] = ss; __syncthreads();
  for (int s = 128; s > 0; s >>= 1) { if (tid < s) red[tid] += red[tid + s]; __syncthreads(); }
  float inv = rsqrtf(red[0] / (float)cols + 1e-6f);
  for (int c = tid; c < cols; c += 256) out[row * (size_t)cols + c] = f2bf(xr[c] * inv * w[c]);
}

__global__ void qk_post_kernel(const float* __restrict__ pre, const float* __restrict__ cw,
                               const float* __restrict__ cb, unsigned short* __restrict__ out) {
  __shared__ float red[256];
  int bt = blockIdx.x;
  int b = bt / TT, t = bt % TT;
  int tid = threadIdx.x;
  float vals[8];
  float ss = 0.f;
#pragma unroll
  for (int u = 0; u < 8; ++u) {
    int c = tid + u * 256;
    float acc = cb[c];
#pragma unroll
    for (int i = 0; i < KSZ; ++i) {
      int tt = t - (KSZ - 1) + i;
      if (tt >= 0) acc += cw[c * KSZ + i] * pre[((size_t)(b * TT + tt)) * HK + c];
    }
    float s = acc / (1.f + __expf(-acc)); // swish
    vals[u] = s; ss += s * s;
  }
  red[tid] = ss; __syncthreads();
  for (int s = 128; s > 0; s >>= 1) { if (tid < s) red[tid] += red[tid + s]; __syncthreads(); }
  float inv = 1.f / (sqrtf(red[0]) + 1e-6f);
#pragma unroll
  for (int u = 0; u < 8; ++u) {
    int c = tid + u * 256;
    out[(size_t)bt * HK + c] = f2bf(vals[u] * inv);
  }
}

__global__ void swish_bf16_kernel(const float* __restrict__ s, unsigned short* __restrict__ d, size_t n) {
  size_t i = (size_t)blockIdx.x * blockDim.x + threadIdx.x;
  if (i < n) { float v = s[i]; d[i] = f2bf(v / (1.f + __expf(-v))); }
}
__global__ void sigmoid_bf16_kernel(const float* __restrict__ s, unsigned short* __restrict__ d, size_t n) {
  size_t i = (size_t)blockIdx.x * blockDim.x + threadIdx.x;
  if (i < n) d[i] = f2bf(1.f / (1.f + __expf(-s[i])));
}
__global__ void sigmoid_f32_kernel(const float* __restrict__ s, float* __restrict__ d, size_t n) {
  size_t i = (size_t)blockIdx.x * blockDim.x + threadIdx.x;
  if (i < n) d[i] = 1.f / (1.f + __expf(-s[i]));
}
__global__ void logsigmoid_bf16_kernel(const float* __restrict__ s, unsigned short* __restrict__ d, size_t n) {
  size_t i = (size_t)blockIdx.x * blockDim.x + threadIdx.x;
  if (i < n) { float x = s[i]; d[i] = f2bf(fminf(x, 0.f) - log1pf(__expf(-fabsf(x)))); }
}
__global__ void add_kernel(const float* __restrict__ a, const float* __restrict__ b,
                           float* __restrict__ d, size_t n) {
  size_t i = (size_t)blockIdx.x * blockDim.x + threadIdx.x;
  if (i < n) d[i] = a[i] + b[i];
}
__global__ void silu_mul_kernel(const unsigned short* __restrict__ t1, const unsigned short* __restrict__ t3,
                                unsigned short* __restrict__ d, size_t n) {
  size_t i = (size_t)blockIdx.x * blockDim.x + threadIdx.x;
  if (i < n) {
    float a = bf2f(t1[i]);
    float s = a / (1.f + __expf(-a));
    d[i] = f2bf(s * bf2f(t3[i]));
  }
}
__global__ void gate_norm_kernel(const unsigned short* __restrict__ o, const float* __restrict__ w,
                                 const float* __restrict__ gate, unsigned short* __restrict__ out) {
  __shared__ float red[128];
  size_t row = blockIdx.x;
  int d = threadIdx.x;
  float v = bf2f(o[row * HD + d]);
  red[d] = v * v; __syncthreads();
  for (int s = 64; s > 0; s >>= 1) { if (d < s) red[d] += red[d + s]; __syncthreads(); }
  float inv = rsqrtf(red[0] / (float)HD + 1e-6f);
  out[row * HD + d] = f2bf(v * inv * w[d] * gate[row * HD + d]);
}

// -------------------- WMMA tiled GEMM: C[M,N] = A[M,K] @ B[K,N] --------------------
#define GBM 128
#define GBN 128
#define GBK 32
template<bool OUT_BF16>
__global__ __launch_bounds__(256) void gemm_bf16_kernel(
    const unsigned short* __restrict__ A, const unsigned short* __restrict__ Bw,
    void* __restrict__ Cout, int M, int N, int K) {
  __shared__ unsigned short As[GBM][GBK];   // 8 KB (filled by async DMA when available)
  __shared__ unsigned short Bs[GBN][GBK];   // 8 KB (stored transposed: [n][k])
  int tid = threadIdx.x;
  int wave = tid >> 5, lane = tid & 31;
  int bm = blockIdx.y * GBM, bn = blockIdx.x * GBN;
  int wm = (wave & 3) * 32;
  int wn = (wave >> 2) * 64;

  v8f acc[2][4];
#pragma unroll
  for (int i = 0; i < 2; ++i)
#pragma unroll
    for (int j = 0; j < 4; ++j) acc[i][j] = (v8f){0.f,0.f,0.f,0.f,0.f,0.f,0.f,0.f};

  for (int k0 = 0; k0 < K; k0 += GBK) {
    { // A tile 128x32 -> LDS (async global->LDS on CDNA5; sync fallback otherwise)
      int r = tid >> 1, c = (tid & 1) * 16;
      int gr = bm + r;
      const unsigned short* src = A + (size_t)gr * K + k0 + c;
      unsigned short* dst = &As[r][c];
      if (gr < M) {
#if HAVE_ASYNC
        __builtin_amdgcn_global_load_async_to_lds_b128(
            (GAS v4i_t*)(const void*)src, (LAS v4i_t*)(void*)dst, 0, 0);
        __builtin_amdgcn_global_load_async_to_lds_b128(
            (GAS v4i_t*)(const void*)(src + 8), (LAS v4i_t*)(void*)(dst + 8), 0, 0);
#else
        *(uint4*)dst       = *(const uint4*)src;
        *(uint4*)(dst + 8) = *(const uint4*)(src + 8);
#endif
        if (k0 + GBK < K) __builtin_prefetch(src + GBK, 0, 1);
      } else {
        uint4 z = {0,0,0,0};
        *(uint4*)dst = z; *(uint4*)(dst + 8) = z;
      }
    }
    { // B tile 32x128 -> transposed into Bs[n][k]
      int kk = tid >> 3, cc = (tid & 7) * 16;
      int gc = bn + cc;
      unsigned short tmp[16];
      uint4 z0 = {0,0,0,0}, z1 = {0,0,0,0};
      if (gc < N) {
        const unsigned short* src = Bw + (size_t)(k0 + kk) * N + gc;
        z0 = *(const uint4*)src;
        z1 = *(const uint4*)(src + 8);
      }
      *(uint4*)tmp       = z0;
      *(uint4*)(tmp + 8) = z1;
#pragma unroll
      for (int t = 0; t < 16; ++t) Bs[cc + t][kk] = tmp[t];
    }
#if HAVE_ASYNC
    __builtin_amdgcn_s_wait_asynccnt(0);
#endif
    __syncthreads();
    v16bf bfrag[4];
#pragma unroll
    for (int nt = 0; nt < 4; ++nt) bfrag[nt] = fragBt_rm(&Bs[wn + nt * 16][0], GBK);
#pragma unroll
    for (int mt = 0; mt < 2; ++mt) {
      v16bf a = fragA_rm(&As[wm + mt * 16][0], GBK);
#pragma unroll
      for (int nt = 0; nt < 4; ++nt)
        acc[mt][nt] = wmma_bf16(a, bfrag[nt], acc[mt][nt]);
    }
    __syncthreads();
  }
#pragma unroll
  for (int mt = 0; mt < 2; ++mt)
#pragma unroll
    for (int nt = 0; nt < 4; ++nt) {
      int mbase = bm + wm + mt * 16 + ((lane >> 4) << 3);
      int n = bn + wn + nt * 16 + (lane & 15);
#pragma unroll
      for (int r = 0; r < 8; ++r) {
        int m = mbase + r;
        if (m < M && n < N) {
          if (OUT_BF16) ((unsigned short*)Cout)[(size_t)m * N + n] = f2bf(acc[mt][nt][r]);
          else          ((float*)Cout)[(size_t)m * N + n] = acc[mt][nt][r];
        }
      }
    }
}

// -------------------- KDA chunked delta-rule scan --------------------
// One workgroup per (b,h): 8 waves, chunk state in LDS (~152 KB of 320 KB/WGP).
// Chunk tiles are fetched by the Tensor Data Mover (2D D#: 64 rows x 256B, stride HK*2B).
__global__ __launch_bounds__(256) void kda_kernel(
    const unsigned short* __restrict__ q, const unsigned short* __restrict__ k,
    const unsigned short* __restrict__ v, const unsigned short* __restrict__ g,
    const unsigned short* __restrict__ beta,
    unsigned short* __restrict__ o, unsigned short* __restrict__ S_out) {
  __shared__ unsigned short Ssh[HD][HD];          // state S[d][v] (bf16, as reference)
  __shared__ unsigned short gcs[CHUNK][HD];       // cumsum(g)
  __shared__ unsigned short qg[CHUNK][HD];        // q * exp(g) * K^-1/2
  __shared__ unsigned short kg[CHUNK][HD];        // k * exp(g)
  __shared__ unsigned short kng[CHUNK][HD];       // k * exp(-g)
  __shared__ unsigned short vs[CHUNK][HD];        // v
  __shared__ unsigned short wsh[CHUNK][HD];       // w = A @ (eg*k)
  __shared__ unsigned short us[CHUNK][HD];        // u then v_i in place
  __shared__ unsigned short Ash[CHUNK][CHUNK];    // A then A_qk
  __shared__ float betas[CHUNK];

  int tid = threadIdx.x;
  int wave = tid >> 5, lane = tid & 31;
  int bh = blockIdx.x;
  int b = bh / HEADS, h = bh % HEADS;
  const float qscale = rsqrtf((float)HD);

  for (int idx = tid; idx < HD * HD; idx += 256) Ssh[idx >> 7][idx & 127] = 0;
  __syncthreads();

  for (int n = 0; n < NCHUNK; ++n) {
    int t0 = n * CHUNK;
    // ---- load chunk tiles into LDS ----
#if HAVE_TDM
    if (wave == 0) {
      const size_t base = ((size_t)(b * TT + t0)) * HK + (size_t)h * HD;
      tdm_load_2d_bf16(q + base, &qg[0][0],  HD, CHUNK, HK);
      tdm_load_2d_bf16(k + base, &kg[0][0],  HD, CHUNK, HK);
      tdm_load_2d_bf16(k + base, &kng[0][0], HD, CHUNK, HK);
      tdm_load_2d_bf16(v + base, &vs[0][0],  HD, CHUNK, HK);
      tdm_load_2d_bf16(g + base, &gcs[0][0], HD, CHUNK, HK);
      __builtin_amdgcn_s_wait_tensorcnt(0);
    }
#else
    for (int idx = tid; idx < CHUNK * HD; idx += 256) {
      int i = idx >> 7, d = idx & 127;
      size_t off = ((size_t)(b * TT + t0 + i)) * HK + (size_t)h * HD + d;
      qg[i][d] = q[off]; kg[i][d] = k[off]; kng[i][d] = k[off];
      vs[i][d] = v[off]; gcs[i][d] = g[off];
    }
#endif
    if (tid < CHUNK) betas[tid] = bf2f(beta[((size_t)(b * TT + t0 + tid)) * HEADS + h]);
    __syncthreads();
    // cumsum g along chunk rows, per feature column
    if (tid < HD) {
      float s = 0.f;
      for (int i = 0; i < CHUNK; ++i) { s += bf2f(gcs[i][tid]); gcs[i][tid] = f2bf(s); }
    }
    __syncthreads();
    // elementwise: qg = q*eg*scale, kg = k*eg, kng = k*eng
    for (int idx = tid; idx < CHUNK * HD; idx += 256) {
      int i = idx >> 7, d = idx & 127;
      float gg = bf2f(gcs[i][d]);
      float eg = __expf(gg), eng = __expf(-gg);
      qg[i][d] = f2bf(bf2f(qg[i][d]) * qscale * eg);
      float kf = bf2f(kg[i][d]);
      kg[i][d]  = f2bf(kf * eg);
      kng[i][d] = f2bf(kf * eng);
    }
    __syncthreads();
    // A[j][i] = kg[j] . kng[i]  (64x64x128), then A = -(tril ? A*beta[j] : 0)
    for (int tix = wave; tix < 16; tix += 8) {
      int mt = tix >> 2, nt = tix & 3;
      v8f acc = (v8f){0.f,0.f,0.f,0.f,0.f,0.f,0.f,0.f};
      for (int kk = 0; kk < HD; kk += 32) {
        v16bf a  = fragA_rm(&kg[mt * 16][kk], HD);
        v16bf bf = fragBt_rm(&kng[nt * 16][kk], HD);
        acc = wmma_bf16(a, bf, acc);
      }
      int jb = mt * 16 + ((lane >> 4) << 3);
      int i  = nt * 16 + (lane & 15);
#pragma unroll
      for (int r = 0; r < 8; ++r) {
        int j = jb + r;
        float bj = betas[j];                 // unconditional load -> select, not branch
        float val = -(acc[r] * bj);
        Ash[j][i] = f2bf((j >= i) ? val : 0.f);
      }
    }
    __syncthreads();
    // forward substitution (column-independent in reference formulation)
    if (tid < CHUNK) {
      int l = tid;
      for (int it = 1; it < CHUNK; ++it) {
        float cs = 0.f;
        for (int j = 0; j < CHUNK; ++j) cs += bf2f(Ash[j][l]);
        float a = bf2f(Ash[it][l]);
        Ash[it][l] = f2bf(a + a * cs);
      }
    }
    __syncthreads();
    // A = (A + I) * beta[i]
    for (int idx = tid; idx < CHUNK * CHUNK; idx += 256) {
      int j = idx >> 6, i = idx & 63;
      float a = bf2f(Ash[j][i]);
      float d = (j == i) ? 1.f : 0.f;
      Ash[j][i] = f2bf((a + d) * betas[i]);
    }
    __syncthreads();
    // w = A @ kg ; u = A @ v   (64x128x64)
    for (int tix = wave; tix < 32; tix += 8) {
      int mt = tix >> 3, nt = tix & 7;
      v8f aw = (v8f){0.f,0.f,0.f,0.f,0.f,0.f,0.f,0.f};
      v8f au = aw;
      for (int kk = 0; kk < CHUNK; kk += 32) {
        v16bf a  = fragA_rm(&Ash[mt * 16][kk], CHUNK);
        v16bf b1 = fragB_kmajor(&kg[kk][0], HD, nt * 16);
        v16bf b2 = fragB_kmajor(&vs[kk][0], HD, nt * 16);
        aw = wmma_bf16(a, b1, aw);
        au = wmma_bf16(a, b2, au);
      }
      int mb = mt * 16 + ((lane >> 4) << 3);
      int d  = nt * 16 + (lane & 15);
#pragma unroll
      for (int r = 0; r < 8; ++r) {
        wsh[mb + r][d] = f2bf(aw[r]);
        us[mb + r][d]  = f2bf(au[r]);
      }
    }
    __syncthreads();
    // A_qk = qg @ kng^T, causal mask (keep j<=i)
    for (int tix = wave; tix < 16; tix += 8) {
      int mt = tix >> 2, nt = tix & 3;
      v8f acc = (v8f){0.f,0.f,0.f,0.f,0.f,0.f,0.f,0.f};
      for (int kk = 0; kk < HD; kk += 32) {
        v16bf a  = fragA_rm(&qg[mt * 16][kk], HD);
        v16bf bf = fragBt_rm(&kng[nt * 16][kk], HD);
        acc = wmma_bf16(a, bf, acc);
      }
      int ib = mt * 16 + ((lane >> 4) << 3);
      int j  = nt * 16 + (lane & 15);
#pragma unroll
      for (int r = 0; r < 8; ++r) {
        int i = ib + r;
        Ash[i][j] = f2bf((j <= i) ? acc[r] : 0.f);
      }
    }
    __syncthreads();
    // v_i = u - w @ S   (64x128x128), in place in us
    for (int tix = wave; tix < 32; tix += 8) {
      int mt = tix >> 3, nt = tix & 7;
      v8f acc = (v8f){0.f,0.f,0.f,0.f,0.f,0.f,0.f,0.f};
      for (int kk = 0; kk < HD; kk += 32) {
        v16bf a  = fragA_rm(&wsh[mt * 16][kk], HD);
        v16bf bf = fragB_kmajor(&Ssh[kk][0], HD, nt * 16);
        acc = wmma_bf16(a, bf, acc);
      }
      int mb = mt * 16 + ((lane >> 4) << 3);
      int d  = nt * 16 + (lane & 15);
#pragma unroll
      for (int r = 0; r < 8; ++r) {
        int i = mb + r;
        us[i][d] = f2bf(bf2f(us[i][d]) - acc[r]);
      }
    }
    __syncthreads();
    // o = qg @ S + A_qk @ v_i -> global
    for (int tix = wave; tix < 32; tix += 8) {
      int mt = tix >> 3, nt = tix & 7;
      v8f acc = (v8f){0.f,0.f,0.f,0.f,0.f,0.f,0.f,0.f};
      for (int kk = 0; kk < HD; kk += 32) {
        v16bf a  = fragA_rm(&qg[mt * 16][kk], HD);
        v16bf bf = fragB_kmajor(&Ssh[kk][0], HD, nt * 16);
        acc = wmma_bf16(a, bf, acc);
      }
      for (int kk = 0; kk < CHUNK; kk += 32) {
        v16bf a  = fragA_rm(&Ash[mt * 16][kk], CHUNK);
        v16bf bf = fragB_kmajor(&us[kk][0], HD, nt * 16);
        acc = wmma_bf16(a, bf, acc);
      }
      int mb = mt * 16 + ((lane >> 4) << 3);
      int d  = nt * 16 + (lane & 15);
#pragma unroll
      for (int r = 0; r < 8; ++r) {
        int i = mb + r;
        o[((size_t)(b * TT + t0 + i)) * HK + (size_t)h * HD + d] = f2bf(acc[r]);
      }
    }
    __syncthreads();
    // S = S * exp(g_last)[d] + kng^T @ v_i   (128x128x64)
    for (int tix = wave; tix < 64; tix += 8) {
      int mt = tix >> 3, nt = tix & 7;
      v8f acc = (v8f){0.f,0.f,0.f,0.f,0.f,0.f,0.f,0.f};
      for (int kk = 0; kk < CHUNK; kk += 32) {
        v16bf a  = fragA_gatherT(&kng[kk][0], HD, mt * 16);
        v16bf bf = fragB_kmajor(&us[kk][0], HD, nt * 16);
        acc = wmma_bf16(a, bf, acc);
      }
      int db = mt * 16 + ((lane >> 4) << 3);
      int vv = nt * 16 + (lane & 15);
#pragma unroll
      for (int r = 0; r < 8; ++r) {
        int d = db + r;
        float dec = __expf(bf2f(gcs[CHUNK - 1][d]));
        Ssh[d][vv] = f2bf(bf2f(Ssh[d][vv]) * dec + acc[r]);
      }
    }
    __syncthreads();
  }
  for (int idx = tid; idx < HD * HD; idx += 256)
    S_out[(size_t)bh * HD * HD + idx] = Ssh[idx >> 7][idx & 127];
}

// -------------------- host orchestration --------------------
static inline size_t align256(size_t x) { return (x + 255) & ~(size_t)255; }

extern "C" void kernel_launch(void* const* d_in, const int* in_sizes, int n_in,
                              void* d_out, int out_size, void* d_ws, size_t ws_size,
                              hipStream_t stream) {
  (void)in_sizes; (void)n_in; (void)out_size; (void)ws_size;
  const float* x        = (const float*)d_in[0];
  const float* rms1_w   = (const float*)d_in[1];
  const float* wq       = (const float*)d_in[2];
  const float* wk       = (const float*)d_in[3];
  const float* wv       = (const float*)d_in[4];
  const float* q_conv_w = (const float*)d_in[5];
  const float* q_conv_b = (const float*)d_in[6];
  const float* k_conv_w = (const float*)d_in[7];
  const float* k_conv_b = (const float*)d_in[8];
  const float* w_beta   = (const float*)d_in[9];
  const float* alpha_dn = (const float*)d_in[10];
  const float* alpha_up = (const float*)d_in[11];
  const float* gate_dn  = (const float*)d_in[12];
  const float* gate_up  = (const float*)d_in[13];
  const float* rms2_w   = (const float*)d_in[14];
  const float* wo       = (const float*)d_in[15];
  const float* rms3_w   = (const float*)d_in[16];
  const float* w1       = (const float*)d_in[17];
  const float* w2       = (const float*)d_in[18];
  const float* w3       = (const float*)d_in[19];

  char* ws = (char*)d_ws;
  size_t cur = 0;
  auto alloc = [&](size_t bytes) { size_t o = cur; cur += align256(bytes); return (void*)(ws + o); };

  const size_t BTD = (size_t)BT * DIMM;
  const size_t BTI = (size_t)BT * INTER;

  unsigned short* xn    = (unsigned short*)alloc(BTD * 2);
  unsigned short* qb    = (unsigned short*)alloc(BTD * 2);
  unsigned short* kb    = (unsigned short*)alloc(BTD * 2);
  unsigned short* vb    = (unsigned short*)alloc(BTD * 2);
  unsigned short* gb    = (unsigned short*)alloc(BTD * 2);
  unsigned short* betab = (unsigned short*)alloc((size_t)BT * HEADS * 2);
  float*          gatef = (float*)alloc(BTD * 4);
  unsigned short* ob    = (unsigned short*)alloc(BTD * 2);
  unsigned short* outb  = (unsigned short*)alloc(BTD * 2);
  float*          ybuf  = (float*)alloc(BTD * 4);
  unsigned short* hb    = (unsigned short*)alloc(BTD * 2);
  unsigned short* t1b   = (unsigned short*)alloc(BTI * 2);
  unsigned short* t3b   = (unsigned short*)alloc(BTI * 2);
  unsigned short* actb  = (unsigned short*)alloc(BTI * 2);
  float*          fscr  = (float*)alloc(BTD * 4);
  unsigned short* midb  = (unsigned short*)alloc((size_t)BT * HD * 2);
  unsigned short* wqb  = (unsigned short*)alloc((size_t)DIMM * HK * 2);
  unsigned short* wkb  = (unsigned short*)alloc((size_t)DIMM * HK * 2);
  unsigned short* wvb  = (unsigned short*)alloc((size_t)DIMM * HK * 2);
  unsigned short* wob  = (unsigned short*)alloc((size_t)HK * DIMM * 2);
  unsigned short* wbb  = (unsigned short*)alloc((size_t)DIMM * HEADS * 2);
  unsigned short* adb  = (unsigned short*)alloc((size_t)DIMM * HD * 2);
  unsigned short* aub  = (unsigned short*)alloc((size_t)HD * HK * 2);
  unsigned short* gdb  = (unsigned short*)alloc((size_t)DIMM * HD * 2);
  unsigned short* gub  = (unsigned short*)alloc((size_t)HD * HK * 2);
  unsigned short* w1b  = (unsigned short*)alloc((size_t)DIMM * INTER * 2);
  unsigned short* w3b  = (unsigned short*)alloc((size_t)DIMM * INTER * 2);
  unsigned short* w2b  = (unsigned short*)alloc((size_t)INTER * DIMM * 2);

  auto cvt = [&](const float* s, unsigned short* d, size_t n) {
    int blk = (int)((n + 255) / 256);
    cvt_f32_bf16_kernel<<<blk, 256, 0, stream>>>(s, d, n);
  };
  cvt(wq, wqb, (size_t)DIMM * HK);     cvt(wk, wkb, (size_t)DIMM * HK);
  cvt(wv, wvb, (size_t)DIMM * HK);     cvt(wo, wob, (size_t)HK * DIMM);
  cvt(w_beta, wbb, (size_t)DIMM * HEADS);
  cvt(alpha_dn, adb, (size_t)DIMM * HD); cvt(alpha_up, aub, (size_t)HD * HK);
  cvt(gate_dn, gdb, (size_t)DIMM * HD);  cvt(gate_up, gub, (size_t)HD * HK);
  cvt(w1, w1b, (size_t)DIMM * INTER);  cvt(w3, w3b, (size_t)DIMM * INTER);
  cvt(w2, w2b, (size_t)INTER * DIMM);

  rmsnorm_kernel<<<BT, 256, 0, stream>>>(x, rms1_w, xn, DIMM);

  dim3 gproj(HK / GBN, BT / GBM);
  size_t nBTD = BTD;
  int eb = (int)((nBTD + 255) / 256);

  gemm_bf16_kernel<false><<<gproj, 256, 0, stream>>>(xn, wqb, fscr, BT, HK, DIMM);
  qk_post_kernel<<<BT, 256, 0, stream>>>(fscr, q_conv_w, q_conv_b, qb);
  gemm_bf16_kernel<false><<<gproj, 256, 0, stream>>>(xn, wkb, fscr, BT, HK, DIMM);
  qk_post_kernel<<<BT, 256, 0, stream>>>(fscr, k_conv_w, k_conv_b, kb);
  gemm_bf16_kernel<false><<<gproj, 256, 0, stream>>>(xn, wvb, fscr, BT, HK, DIMM);
  swish_bf16_kernel<<<eb, 256, 0, stream>>>(fscr, vb, nBTD);
  gemm_bf16_kernel<false><<<dim3(1, BT / GBM), 256, 0, stream>>>(xn, wbb, fscr, BT, HEADS, DIMM);
  sigmoid_bf16_kernel<<<(int)(((size_t)BT * HEADS + 255) / 256), 256, 0, stream>>>(fscr, betab, (size_t)BT * HEADS);
  gemm_bf16_kernel<true><<<dim3(1, BT / GBM), 256, 0, stream>>>(xn, adb, midb, BT, HD, DIMM);
  gemm_bf16_kernel<false><<<gproj, 256, 0, stream>>>(midb, aub, fscr, BT, HK, HD);
  logsigmoid_bf16_kernel<<<eb, 256, 0, stream>>>(fscr, gb, nBTD);
  gemm_bf16_kernel<true><<<dim3(1, BT / GBM), 256, 0, stream>>>(xn, gdb, midb, BT, HD, DIMM);
  gemm_bf16_kernel<false><<<gproj, 256, 0, stream>>>(midb, gub, fscr, BT, HK, HD);
  sigmoid_f32_kernel<<<eb, 256, 0, stream>>>(fscr, gatef, nBTD);

  unsigned short* S_out = (unsigned short*)((char*)d_out + BTD * sizeof(float));
  kda_kernel<<<BB * HEADS, 256, 0, stream>>>(qb, kb, vb, gb, betab, ob, S_out);

  gate_norm_kernel<<<BT * HEADS, HD, 0, stream>>>(ob, rms2_w, gatef, outb);

  gemm_bf16_kernel<false><<<gproj, 256, 0, stream>>>(outb, wob, fscr, BT, DIMM, HK);
  add_kernel<<<eb, 256, 0, stream>>>(x, fscr, ybuf, nBTD);

  rmsnorm_kernel<<<BT, 256, 0, stream>>>(ybuf, rms3_w, hb, DIMM);
  dim3 gmlp(INTER / GBN, BT / GBM);
  gemm_bf16_kernel<true><<<gmlp, 256, 0, stream>>>(hb, w1b, t1b, BT, INTER, DIMM);
  gemm_bf16_kernel<true><<<gmlp, 256, 0, stream>>>(hb, w3b, t3b, BT, INTER, DIMM);
  silu_mul_kernel<<<(int)((BTI + 255) / 256), 256, 0, stream>>>(t1b, t3b, actb, BTI);
  gemm_bf16_kernel<false><<<gproj, 256, 0, stream>>>(actb, w2b, fscr, BT, DIMM, INTER);

  add_kernel<<<eb, 256, 0, stream>>>(ybuf, fscr, (float*)d_out, nBTD);
}
